// MyRNN_35631048687674
// MI455X (gfx1250) — compile-verified, matrix operand order checked by reading, not run on hardware
//
#include <hip/hip_runtime.h>
#include <hip/hip_bf16.h>
#include <math.h>

// ---------------- problem sizes ----------------
#define TLEN 32768
#define ISZ  1024
#define HSZ  512
#define OSZ  128

typedef __attribute__((ext_vector_type(2)))  float   v2f;
typedef __attribute__((ext_vector_type(8)))  float   v8f;
typedef __attribute__((ext_vector_type(16))) __bf16  v16bf;
typedef __attribute__((ext_vector_type(8)))  __bf16  v8bf;

// Async copy global -> LDS (16B per lane), ASYNCcnt-tracked (gfx1250).
// LDS operand: low 32 bits of the flat shared address == wave LDS offset.
__device__ __forceinline__ void async_copy_b128(void* lds_ptr, const float* gptr) {
    const uint32_t lds_off = (uint32_t)(uintptr_t)lds_ptr;
    asm volatile("global_load_async_to_lds_b128 %0, %1, off"
                 :: "v"(lds_off), "v"((uint64_t)(uintptr_t)gptr)
                 : "memory");
}
__device__ __forceinline__ void wait_async0() {
    asm volatile("s_wait_asynccnt 0x0" ::: "memory");
}

// =====================================================================
// Phase 1: xb[T][512] = x @ W1^T + b1   (fp32 WMMA 16x16x4, memory-bound)
// Block: 256 threads (8 waves). Tile: 128 (M) x 64 (N). K chunked by 32,
// double-buffered in LDS via GLOBAL_LOAD_ASYNC_TO_LDS_B128 (ASYNCcnt).
// LDS rows padded to 36 floats (144B): 16B-aligned and bank-conflict-free
// for the strided v2f fragment reads.
// =====================================================================
#define KC   32
#define APAD 36
#define NCHUNK (ISZ / KC)

__global__ __launch_bounds__(256) void rnn_phase1(
    const float* __restrict__ X,   // [T][1024]
    const float* __restrict__ W1,  // [512][1024]
    const float* __restrict__ B1,  // [512]
    float* __restrict__ XB)        // [T][512]
{
    __shared__ alignas(16) float As[2][128 * APAD]; // x chunk, row-major [m][k]
    __shared__ alignas(16) float Bs[2][64  * APAD]; // W1 chunk, row-major [n][k]

    const int tid  = threadIdx.x;
    const int lane = tid & 31;
    const int wave = tid >> 5;
    const int m0   = blockIdx.x * 128;
    const int n0   = blockIdx.y * 64;

    v8f acc[4] = {v8f{}, v8f{}, v8f{}, v8f{}};

    const int mrow = wave * 16 + (lane & 15); // this wave's A row for its frag lane
    const int khi  = (lane >> 4) * 2;         // f32 A/B frag: upper lanes hold K+2,K+3

    // per-thread staging coordinates (A: 128 rows x 32 f32; B: 64 rows x 32 f32)
    const int rA = tid >> 1, cA = (tid & 1) * 16; // 4 x b128 per thread
    const int rB = tid >> 2, cB = (tid & 3) * 8;  // 2 x b128 per thread
    const float* gA = X  + (size_t)(m0 + rA) * ISZ + cA;
    const float* gB = W1 + (size_t)(n0 + rB) * ISZ + cB;

    // ---- prologue: async-stage chunk 0 into buffer 0 ----
    {
#pragma unroll
        for (int q = 0; q < 4; ++q)
            async_copy_b128(&As[0][rA * APAD + cA + 4 * q], gA + 4 * q);
#pragma unroll
        for (int q = 0; q < 2; ++q)
            async_copy_b128(&Bs[0][rB * APAD + cB + 4 * q], gB + 4 * q);
    }

    for (int chunk = 0; chunk < NCHUNK; ++chunk) {
        // wait for this buffer's async copies, make them visible to all waves
        wait_async0();
        __syncthreads();

        // issue next chunk's copies into the other buffer (overlaps compute)
        if (chunk + 1 < NCHUNK) {
            const int nb = (chunk + 1) & 1;
            gA += KC; gB += KC;
#pragma unroll
            for (int q = 0; q < 4; ++q)
                async_copy_b128(&As[nb][rA * APAD + cA + 4 * q], gA + 4 * q);
#pragma unroll
            for (int q = 0; q < 2; ++q)
                async_copy_b128(&Bs[nb][rB * APAD + cB + 4 * q], gB + 4 * q);
        }

        // ---- compute on current buffer: 8 k-steps x 4 N-subtiles = 32 WMMAs ----
        const float* Ab = As[chunk & 1];
        const float* Bb = Bs[chunk & 1];
        for (int kk = 0; kk < KC; kk += 4) {
            v2f a = *(const v2f*)(Ab + mrow * APAD + kk + khi);
#pragma unroll
            for (int nt = 0; nt < 4; ++nt) {
                const int nn = nt * 16 + (lane & 15);
                v2f b = *(const v2f*)(Bb + nn * APAD + kk + khi);
                acc[nt] = __builtin_amdgcn_wmma_f32_16x16x4_f32(
                    false, a, false, b, (short)0, acc[nt], false, false);
            }
        }
        // no trailing barrier needed: the next overwrite of this buffer happens
        // only after the wait+barrier at the top of iteration chunk+1.
    }

    // ---- epilogue: add bias, store. C layout: VGPR v -> (M=v | v+8, N=lane&15) ----
    const int nlo = lane & 15;
    const int hi  = lane >> 4;
#pragma unroll
    for (int nt = 0; nt < 4; ++nt) {
        const int col  = n0 + nt * 16 + nlo;
        const float bv = B1[col];
#pragma unroll
        for (int v = 0; v < 8; ++v) {
            const int row = m0 + wave * 16 + v + 8 * hi;
            XB[(size_t)row * HSZ + col] = acc[nt][v] + bv;
        }
    }
}

// =====================================================================
// Phase 2+3: sequential scan on ONE WGP-resident workgroup (1024 thr, 32 waves).
// W2 held in VGPRs as pre-swizzled bf16 WMMA A-fragments (16 frags x 8 VGPRs).
// Per step: 16x V_WMMA_F32_16X16X32_BF16 per wave; B = bf16(h) broadcast to all
// 16 columns (matvec on the matrix pipe); column n=0 extracted; tanh elementwise;
// next xb_t software-pipelined + global_prefetch.
// =====================================================================
__global__ __launch_bounds__(1024) void rnn_scan(
    const float* __restrict__ XB,  // [T][512]
    const float* __restrict__ W2,  // [512][512]
    const float* __restrict__ B2,  // [512]
    const float* __restrict__ W3,  // [128][512]
    const float* __restrict__ B3,  // [128]
    float* __restrict__ OUT)       // [128]
{
    __shared__ float ys[HSZ];
    __shared__ float hs[HSZ];
    __shared__ alignas(16) __bf16 hbf[HSZ];

    const int tid  = threadIdx.x;
    const int lane = tid & 31;
    const int wave = tid >> 5;          // wave w owns output rows [16w, 16w+16)

    // ---- preload W2 as bf16 A-fragments (ISA 16-bit A 16x32 layout) ----
    // lane<16: m=lane,   K = kb+{0..7} and kb+{16..23} with kb = kt*32
    // lane>=16: m=lane-16, same but kb += 8
    v16bf aW[16];
    {
        const int m  = wave * 16 + (lane & 15);
        const int kb = (lane >> 4) * 8;
#pragma unroll
        for (int kt = 0; kt < 16; ++kt) {
            const float* src = W2 + (size_t)m * HSZ + kt * 32 + kb;
#pragma unroll
            for (int i = 0; i < 8; ++i) {
                const int ko = 2 * i + ((i >= 4) ? 8 : 0);
                aW[kt][2 * i]     = (__bf16)src[ko];
                aW[kt][2 * i + 1] = (__bf16)src[ko + 1];
            }
        }
    }

    // ---- h0 = 0; pipeline first xb row ----
    if (tid < HSZ) { hs[tid] = 0.0f; hbf[tid] = (__bf16)0.0f; }
    float xv  = (tid < HSZ) ? XB[tid] : 0.0f;
    float b2v = (tid < HSZ) ? B2[tid] : 0.0f;
    __syncthreads();

    const int kb = (lane >> 4) * 8;

    for (int t = 0; t < TLEN; ++t) {
        // ---- y = W2 @ h via 16 chained bf16 WMMAs (f32 accumulate) ----
        v8f acc = v8f{};
#pragma unroll
        for (int kt = 0; kt < 16; ++kt) {
            // B fragment: column-independent broadcast of h chunk (same lane K
            // layout as A, role-mirrored): regs 0..3 = K kb..kb+7, regs 4..7 = +16.
            const v8bf lo = *(const v8bf*)(hbf + kt * 32 + kb);
            const v8bf hi = *(const v8bf*)(hbf + kt * 32 + kb + 16);
            v16bf bfr;
#pragma unroll
            for (int i = 0; i < 8; ++i) { bfr[i] = lo[i]; bfr[8 + i] = hi[i]; }
            acc = __builtin_amdgcn_wmma_f32_16x16x32_bf16(
                false, aW[kt], false, bfr, (short)0, acc, false, false);
        }
        // ---- extract column n=0: lane0 -> M=v, lane16 -> M=v+8 ----
        if ((lane & 15) == 0) {
            const int mbase = wave * 16 + (lane >> 4) * 8;
#pragma unroll
            for (int v = 0; v < 8; ++v) ys[mbase + v] = acc[v];
        }
        __syncthreads();
        // ---- h = tanh(xb_t + y + b2); pipeline next xb_t ----
        if (tid < HSZ) {
            const float hn = tanhf(xv + ys[tid] + b2v);
            hs[tid]  = hn;
            hbf[tid] = (__bf16)hn;
            if (t + 1 < TLEN) xv = XB[(size_t)(t + 1) * HSZ + tid];
        }
        if (tid == 0 && t + 8 < TLEN)
            __builtin_prefetch(XB + (size_t)(t + 8) * HSZ, 0, 0);
        __syncthreads();
    }

    // ---- out = W3 @ h_final + b3 (tiny: 128 x 512 matvec) ----
    if (tid < OSZ) {
        float s = B3[tid];
        const float* w = W3 + (size_t)tid * HSZ;
        for (int k = 0; k < HSZ; ++k) s += w[k] * hs[k];
        OUT[tid] = s;
    }
}

// =====================================================================
extern "C" void kernel_launch(void* const* d_in, const int* in_sizes, int n_in,
                              void* d_out, int out_size, void* d_ws, size_t ws_size,
                              hipStream_t stream) {
    const float* x  = (const float*)d_in[0];
    const float* W1 = (const float*)d_in[1];
    const float* b1 = (const float*)d_in[2];
    const float* W2 = (const float*)d_in[3];
    const float* b2 = (const float*)d_in[4];
    const float* W3 = (const float*)d_in[5];
    const float* b3 = (const float*)d_in[6];
    float* xb  = (float*)d_ws;          // T*H*4 = 64 MB scratch
    float* out = (float*)d_out;

    dim3 grid1(TLEN / 128, HSZ / 64);   // 256 x 8 blocks
    rnn_phase1<<<grid1, 256, 0, stream>>>(x, W1, b1, xb);
    rnn_scan<<<1, 1024, 0, stream>>>(xb, W2, b2, W3, b3, out);
}
